// lstm_parallel_3032246911583
// MI455X (gfx1250) — matrix operand order repaired and checked by reading, not compiled, not run on hardware
//
#include <hip/hip_runtime.h>
#include <hip/hip_bf16.h>
#include <math.h>

typedef __bf16 v16bf __attribute__((ext_vector_type(16)));
typedef float  v8f   __attribute__((ext_vector_type(8)));
typedef unsigned int v4u __attribute__((ext_vector_type(4)));
typedef int    v4i   __attribute__((ext_vector_type(4)));
typedef int    v8i   __attribute__((ext_vector_type(8)));

#define B_   128
#define T_   512
#define D_   64
#define H_   256
#define G4_  1024   // 4*H

#if defined(__has_builtin)
#if __has_builtin(__builtin_amdgcn_tensor_load_to_lds) && __has_builtin(__builtin_amdgcn_s_wait_tensorcnt)
#define USE_TDM 1
#else
#define USE_TDM 0
#endif
#else
#define USE_TDM 0
#endif

// Fast activations on the hardware transcendental unit:
//   v_exp_f32 + v_rcp_f32, ~3-5 VALU ops instead of libm's ~25.
__device__ __forceinline__ float sigf(float x) {
    return __builtin_amdgcn_rcpf(1.0f + __expf(-x));       // x->+inf: 1 ; x->-inf: 0
}
__device__ __forceinline__ float tanhfast(float x) {
    return 1.0f - 2.0f * __builtin_amdgcn_rcpf(1.0f + __expf(2.0f * x)); // saturates to ±1
}

// ---------------- workspace layout (bytes) ----------------
// hcontrib : 128*1024 f32 @ 0        | wihbf : 1024*64 bf16 @ 524288
// hv0/hv1/cv0/cv1 : 128*256 f32 @ 655360/786432/917504/1048576
// s0 : 128 f32 @ 1179648 | s1 : 128 f32 @ 1180160

__global__ void k_zero(float* acc, int n) {
    int i = blockIdx.x * blockDim.x + threadIdx.x;
    if (i < n) acc[i] = 0.0f;
}

__global__ void k_cvt_wih(const float* __restrict__ wih, __bf16* __restrict__ out) {
    int i = blockIdx.x * blockDim.x + threadIdx.x;   // 65536 threads
    out[i] = (__bf16)wih[i];
}

// hcontrib[b,g] = b_ih[g] + b_hh[g] + sum_h h0[b,h]*W_hh[g,h]
__global__ void k_hcontrib(const float* __restrict__ h0, const float* __restrict__ whh,
                           const float* __restrict__ bih, const float* __restrict__ bhh,
                           float* __restrict__ hc) {
    int idx = blockIdx.x * blockDim.x + threadIdx.x; // 131072 threads
    int b = idx >> 10;
    int g = idx & 1023;
    const float* hr = h0  + b * H_;
    const float* wr = whh + (size_t)g * H_;
    float a = bih[g] + bhh[g];
#pragma unroll 8
    for (int h = 0; h < H_; ++h) a += hr[h] * wr[h];
    hc[idx] = a;
}

// Fused: gates GEMM (WMMA bf16) + activations + weighted T-reduction.
// grid = 512 blocks (128 b * 4 t-chunks of 128), 256 threads = 8 waves.
// dynamic LDS: 131072 B (W_ih bf16) + 4096 B (per-block accumulators)
__global__ void k_main(const float* __restrict__ x, const float* __restrict__ c0,
                       const float* __restrict__ hcontrib, const __bf16* __restrict__ wihbf,
                       float* __restrict__ hv0, float* __restrict__ hv1,
                       float* __restrict__ cv0, float* __restrict__ cv1,
                       float* __restrict__ s0, float* __restrict__ s1,
                       const int* __restrict__ longp) {
    extern __shared__ char smem[];
    __bf16* lds_w   = (__bf16*)smem;              // 1024*64 bf16
    float*  lds_acc = (float*)(smem + 131072);    // [4][256]

    const int b     = blockIdx.x >> 2;
    const int tbase = (blockIdx.x & 3) * 128;
    const int tid   = threadIdx.x;
    const int lane  = tid & 31;
    const int wave  = tid >> 5;
    const float invLong = 1.0f / (float)(*longp);

#if USE_TDM
    // Stage W_ih (bf16, 1024x64, 128 KB) into LDS with one Tensor-Data-Mover DMA.
    // D# per CDNA5 ISA ch.8: 2D tensor, data_size=2B, tile == whole tensor.
    if (wave == 0) {
        unsigned lds_base = (unsigned)__builtin_amdgcn_groupstaticsize(); // dynamic LDS starts here
        unsigned long long ga = (unsigned long long)(uintptr_t)wihbf;
        v4u g0;
        g0[0] = 1u;                                              // count=1 (valid user D#)
        g0[1] = lds_base;                                        // lds_addr[31:0]
        g0[2] = (unsigned)(ga & 0xffffffffu);                    // global_addr[31:0]
        g0[3] = (unsigned)((ga >> 32) & 0x01ffffffu) | (2u << 30); // global_addr[56:32] | type=2
        v8i g1;
        g1[0] = 0x00010000;            // workgroup_mask=0, data_size=1 (2 bytes)
        g1[1] = (int)(64u << 16);      // tensor_dim0 = 64  (bits[79:48] lo half)
        g1[2] = (int)(1024u << 16);    // tensor_dim1 = 1024 (bits[111:80] lo half)
        g1[3] = (int)(64u << 16);      // tile_dim0 = 64 (bits[127:112])
        g1[4] = 1024;                  // tile_dim1 = 1024 (bits[143:128]); tile_dim2=0
        g1[5] = 64;                    // tensor_dim0_stride = 64 (bits[207:160] lo)
        g1[6] = 0;
        g1[7] = 0;
        v4i gz = {0, 0, 0, 0};
#if defined(__clang_major__) && __clang_major__ >= 23
        v8i gz8 = {0, 0, 0, 0, 0, 0, 0, 0};
        __builtin_amdgcn_tensor_load_to_lds(g0, g1, gz, gz, gz8, 0);
#else
        __builtin_amdgcn_tensor_load_to_lds(g0, g1, gz, gz, 0);
#endif
        __builtin_amdgcn_s_wait_tensorcnt((short)0);
    }
#else
    {
        const uint4* src = (const uint4*)wihbf;
        uint4*       dst = (uint4*)lds_w;
        for (int i = tid; i < 8192; i += 256) dst[i] = src[i];
    }
#endif
    for (int i = tid; i < 1024; i += 256) lds_acc[i] = 0.0f;
    __syncthreads();

    // s0/s1 = sum over ALL t (t=0 included; the mask applies only to h/c).
    if (tid < 128) {
        int t = tbase + tid;
        const float* xr = x + ((size_t)b * T_ + t) * D_;
        float w0 = xr[0] * invLong;
        float w1 = xr[1];
#pragma unroll
        for (int off = 16; off > 0; off >>= 1) {
            w0 += __shfl_down(w0, off);
            w1 += __shfl_down(w1, off);
        }
        if (lane == 0) { atomicAdd(&s0[b], w0); atomicAdd(&s1[b], w1); }
    }

    // ---- A operand: this wave's 16-row t-strip, K=64 split into two K=32 halves.
    // A layout used here (k-contiguous, as B): lanes 0-15 row=lane, K 0..15; lanes 16-31 same row, K 16..31.
    const int tstrip = tbase + wave * 16;
    const int rowA   = tstrip + (lane & 15);
    const int klo    = (lane < 16) ? 0 : 16;
    const float* xa  = x + ((size_t)b * T_ + rowA) * D_;
    v16bf a0, a1;
#pragma unroll
    for (int j = 0; j < 8; ++j) {
        a0[j]     = (__bf16)xa[klo + j];
        a0[8 + j] = (__bf16)xa[klo + 8 + j];
        a1[j]     = (__bf16)xa[32 + klo + j];
        a1[8 + j] = (__bf16)xa[40 + klo + j];
    }

    // Per-lane row weights for the C/D layout (lanes 0-15: m=r, lanes 16-31: m=r+8),
    // with the t>0 mask folded in (masks h/c contributions only).
    float wr0[8], wr1[8];
    const int mbase = (lane < 16) ? 0 : 8;
#pragma unroll
    for (int r = 0; r < 8; ++r) {
        int t = tstrip + mbase + r;
        const float* xw = x + ((size_t)b * T_ + t) * D_;
        float msk = (t > 0) ? 1.0f : 0.0f;
        wr0[r] = msk * xw[0] * invLong;
        wr1[r] = msk * xw[1];
    }

    const int ncol = lane & 15;
    for (int grp = 0; grp < 16; ++grp) {
        const int hcol    = grp * 16 + ncol;
        const float cprev = c0[b * H_ + hcol];

        v8f acc[4];
#pragma unroll
        for (int gate = 0; gate < 4; ++gate) {
            const int col  = gate * H_ + hcol;
            const float hc = hcontrib[b * G4_ + col];
            v8f c;
#pragma unroll
            for (int r = 0; r < 8; ++r) c[r] = hc;
            // B layout used here (as A): lanes 0-15 col=lane K=k0..k0+15; lanes 16-31 col=lane-16 K=k0+16..k0+31.
            const __bf16* wp0 = &lds_w[(size_t)col * D_ + ((lane < 16) ? 0 : 16)];
            const __bf16* wp1 = wp0 + 32;
            v16bf b0, b1;
#pragma unroll
            for (int j = 0; j < 16; ++j) { b0[j] = wp0[j]; b1[j] = wp1[j]; }
            c = __builtin_amdgcn_wmma_f32_16x16x32_bf16(false, a0, false, b0, (short)0, c, false, false);
            c = __builtin_amdgcn_wmma_f32_16x16x32_bf16(false, a1, false, b1, (short)0, c, false, false);
            acc[gate] = c;
        }

        // Activations + LSTM recombination + weighted T-partials, all in-register.
        float ph0 = 0.f, ph1 = 0.f, pc0 = 0.f, pc1 = 0.f;
#pragma unroll
        for (int r = 0; r < 8; ++r) {
            float ig = sigf(acc[0][r]);
            float fg = sigf(acc[1][r]);
            float gg = tanhfast(acc[2][r]);
            float og = sigf(acc[3][r]);
            float cval = fg * cprev + ig * gg;
            float hval = og * tanhfast(cval);
            ph0 += hval * wr0[r];  ph1 += hval * wr1[r];
            pc0 += cval * wr0[r];  pc1 += cval * wr1[r];
        }
        // fold the two row-halves (lane L and L+16 hold the same column)
        ph0 += __shfl_xor(ph0, 16);  ph1 += __shfl_xor(ph1, 16);
        pc0 += __shfl_xor(pc0, 16);  pc1 += __shfl_xor(pc1, 16);
        if (lane < 16) {
            atomicAdd(&lds_acc[0 * H_ + hcol], ph0);
            atomicAdd(&lds_acc[1 * H_ + hcol], ph1);
            atomicAdd(&lds_acc[2 * H_ + hcol], pc0);
            atomicAdd(&lds_acc[3 * H_ + hcol], pc1);
        }
    }

    __syncthreads();
    {   // one global atomic per (b,h) per block
        int i = tid;  // 0..255
        atomicAdd(&hv0[b * H_ + i], lds_acc[0 * H_ + i]);
        atomicAdd(&hv1[b * H_ + i], lds_acc[1 * H_ + i]);
        atomicAdd(&cv0[b * H_ + i], lds_acc[2 * H_ + i]);
        atomicAdd(&cv1[b * H_ + i], lds_acc[3 * H_ + i]);
    }
}

// out[part(h=0,c=1)][b][k] : k<128 -> hv/cv0 @ f1_w + f1_b*s0 ; k>=128 -> hv/cv1 @ f2_w + f2_b*s1
__global__ void k_final(const float* __restrict__ hv0, const float* __restrict__ hv1,
                        const float* __restrict__ cv0, const float* __restrict__ cv1,
                        const float* __restrict__ s0, const float* __restrict__ s1,
                        const float* __restrict__ f1w, const float* __restrict__ f1b,
                        const float* __restrict__ f2w, const float* __restrict__ f2b,
                        float* __restrict__ out) {
    int idx = blockIdx.x * blockDim.x + threadIdx.x;   // 65536
    int part = idx >> 15;
    int rem  = idx & 32767;
    int b    = rem >> 8;
    int k    = rem & 255;

    const float* vec;
    const float* w;
    float bias;
    if (k < 128) {
        vec  = (part ? cv0 : hv0) + b * H_;
        w    = f1w + (size_t)k * H_;
        bias = f1b[k] * s0[b];
    } else {
        int kk = k - 128;
        vec  = (part ? cv1 : hv1) + b * H_;
        w    = f2w + (size_t)kk * H_;
        bias = f2b[kk] * s1[b];
    }
    float a = bias;
#pragma unroll 8
    for (int h = 0; h < H_; ++h) a += vec[h] * w[h];
    out[idx] = a;
}

extern "C" void kernel_launch(void* const* d_in, const int* in_sizes, int n_in,
                              void* d_out, int out_size, void* d_ws, size_t ws_size,
                              hipStream_t stream) {
    const float* x    = (const float*)d_in[0];
    const float* h0   = (const float*)d_in[1];
    const float* c0   = (const float*)d_in[2];
    const float* Wih  = (const float*)d_in[3];
    const float* Whh  = (const float*)d_in[4];
    const float* bih  = (const float*)d_in[5];
    const float* bhh  = (const float*)d_in[6];
    const float* f1w  = (const float*)d_in[7];
    const float* f1b  = (const float*)d_in[8];
    const float* f2w  = (const float*)d_in[9];
    const float* f2b  = (const float*)d_in[10];
    const int*   lng  = (const int*)d_in[11];

    char* ws = (char*)d_ws;
    float*  hcontrib = (float*)(ws + 0);
    __bf16* wihbf    = (__bf16*)(ws + 524288);
    float*  hv0      = (float*)(ws + 655360);
    float*  hv1      = (float*)(ws + 786432);
    float*  cv0      = (float*)(ws + 917504);
    float*  cv1      = (float*)(ws + 1048576);
    float*  s0       = (float*)(ws + 1179648);
    float*  s1       = (float*)(ws + 1180160);

    k_zero<<<(4 * 32768 + 255) / 256, 256, 0, stream>>>(hv0, 4 * 32768);
    k_zero<<<1, 256, 0, stream>>>(s0, 256);  // s0 and s1 are contiguous (128+128)

    k_cvt_wih<<<65536 / 256, 256, 0, stream>>>(Wih, wihbf);
    k_hcontrib<<<131072 / 256, 256, 0, stream>>>(h0, Whh, bih, bhh, hcontrib);

    const size_t lds_bytes = 131072 + 4096;
    k_main<<<512, 256, lds_bytes, stream>>>(x, c0, hcontrib, wihbf,
                                            hv0, hv1, cv0, cv1, s0, s1, lng);

    k_final<<<65536 / 256, 256, 0, stream>>>(hv0, hv1, cv0, cv1, s0, s1,
                                             f1w, f1b, f2w, f2b, (float*)d_out);
}
